// MomentumFormerLayer_32890859553319
// MI455X (gfx1250) — matrix operand order repaired and checked by reading, not compiled, not run on hardware
//
#include <hip/hip_runtime.h>
#include <hip/hip_bf16.h>
#include <math.h>

// ---------------------------------------------------------------------------
// MomentumFormerLayer, fused for MI455X (gfx1250, wave32, WMMA bf16).
// N=16 P=128 CIN=64 C=128 HID=32.  Heavy per-pair GEMMs use
// v_wmma_f32_16x16x32_bf16; softmax over j is done online (flash style) so
// the (N,P,P,C) intermediates never touch HBM — U (134MB) is read once,
// double-buffered through LDS with GLOBAL_LOAD_ASYNC_TO_LDS_B128 (ASYNCcnt).
// ---------------------------------------------------------------------------

typedef __bf16 bf16_t;
typedef __attribute__((ext_vector_type(16))) __bf16 v16bf;
typedef __attribute__((ext_vector_type(8)))  float  v8f;

#define NB   16
#define PP   128
#define CIN  64
#define CD   128
#define HIDD 32
#define NEGV (-10000.0f)

#if __has_builtin(__builtin_amdgcn_global_load_async_to_lds_b128)
#define HAVE_ASYNC_LDS 1
// Builtin signature (from hipcc diagnostic): param0 is a pointer to
// __attribute__((vector_size(16))) int in the global (__device__/AS1) space,
// param1 the LDS (AS3) counterpart, then imm offset + cpol.
typedef int v4i_gcc __attribute__((vector_size(4 * sizeof(int))));
typedef __attribute__((address_space(1))) v4i_gcc* gp128_t;
typedef __attribute__((address_space(3))) v4i_gcc* lp128_t;
#if __has_builtin(__builtin_amdgcn_s_wait_asynccnt)
#define WAIT_ASYNC(n) __builtin_amdgcn_s_wait_asynccnt(n)
#else
#define WAIT_ASYNC(n) asm volatile("s_wait_asynccnt %0" ::"i"(n) : "memory")
#endif
#endif

// ---------------------------------------------------------------------------
// WMMA helper: Out(16 x NOUT, f32, ld=ldo) = A(16 x KTOT bf16, ld=lda)
//              @ W(NOUT x KTOT bf16 row-major, ld=ldw)^T
// A rows = M, W rows = output channel.  4 waves split the N-tiles.
// Per-lane layouts follow CDNA5 ISA 7.12.2 (wave32); the chosen K mapping
// makes both fragments contiguous 16B LDS reads (ds_load_b128).
// ---------------------------------------------------------------------------
template<int KTOT, int NOUT>
__device__ inline void wmma_gemm(const bf16_t* A, int lda,
                                 const bf16_t* W, int ldw,
                                 float* Out, int ldo) {
  const int wave = threadIdx.x >> 5;
  const int lane = threadIdx.x & 31;
  const int half = lane >> 4;
  const int m16  = lane & 15;
  for (int nt = wave; nt < NOUT / 16; nt += 4) {
    v8f acc = {};
#pragma unroll
    for (int kk = 0; kk < KTOT; kk += 32) {
      v16bf a, b;
#pragma unroll
      for (int e = 0; e < 16; ++e) {
        int kA = half * 16 + e;
        a[e] = A[m16 * lda + kk + kA];
      }
      const bf16_t* wrow = W + (nt * 16 + m16) * ldw + kk + half * 16;
#pragma unroll
      for (int e = 0; e < 16; ++e) b[e] = wrow[e];
      acc = __builtin_amdgcn_wmma_f32_16x16x32_bf16(
          false, a, false, b, (short)0, acc, false, false);
    }
#pragma unroll
    for (int v = 0; v < 8; ++v)
      Out[(v + 8 * half) * ldo + nt * 16 + m16] = acc[v];
  }
}

// Row-wise LayerNorm stats for a 16x128 f32 tile (128 threads: 8 per row).
__device__ inline void ln_stats_16x128(const float* X, float* red, float* red2,
                                       float* mean, float* inv) {
  int t = threadIdx.x, r = t >> 3, g8 = t & 7;
  float s = 0.f, s2 = 0.f;
#pragma unroll
  for (int e = 0; e < 16; ++e) {
    float xv = X[r * 128 + g8 * 16 + e];
    s += xv; s2 += xv * xv;
  }
  red[r * 8 + g8] = s; red2[r * 8 + g8] = s2;
  __syncthreads();
  if (t < 16) {
    float a = 0.f, b = 0.f;
#pragma unroll
    for (int e = 0; e < 8; ++e) { a += red[t * 8 + e]; b += red2[t * 8 + e]; }
    float m = a * (1.f / 128.f);
    float v = b * (1.f / 128.f) - m * m;
    mean[t] = m;
    inv[t]  = rsqrtf(v + 1e-5f);
  }
  __syncthreads();
}

// ---------------------------------------------------------------------------
// Kernel 1: q/k/v projections (x @ w.T + b), 2048 rows of (64 -> 128)x3.
// ---------------------------------------------------------------------------
__global__ void qkv_kernel(const float* __restrict__ x,
                           const float* __restrict__ wq, const float* __restrict__ bq,
                           const float* __restrict__ wk, const float* __restrict__ bk,
                           const float* __restrict__ wv, const float* __restrict__ bv,
                           float* __restrict__ q, float* __restrict__ k,
                           float* __restrict__ v) {
  __shared__ float sX[CIN];
  int row = blockIdx.x, t = threadIdx.x;
  if (t < CIN) sX[t] = x[row * CIN + t];
  __syncthreads();
  float aq = bq[t], ak = bk[t], av = bv[t];
#pragma unroll 8
  for (int c = 0; c < CIN; ++c) {
    float xv = sX[c];
    aq += xv * wq[t * CIN + c];
    ak += xv * wk[t * CIN + c];
    av += xv * wv[t * CIN + c];
  }
  q[row * CD + t] = aq; k[row * CD + t] = ak; v[row * CD + t] = av;
}

// ---------------------------------------------------------------------------
// Kernel 2: fused per-(n,i) attention pipeline with online softmax over j and
// async double-buffered U tiles.  Dynamic LDS:
//   sW1 bf16[16384] | sW2 bf16[16384] | sPe2 bf16[4096] | sA bf16[2048]
//   sT1 f32[2048] | sU0 f32[2048] | sU1 f32[2048] | 6x param f32[128]
//   sKi f32[128] | sRed/sRed2 f32[128] | sMean/sInv f32[16] | sPi f32[8]
// ---------------------------------------------------------------------------
__global__ void __launch_bounds__(128)
main_kernel(const float* __restrict__ p, const float* __restrict__ U,
            const float* __restrict__ q_ws, const float* __restrict__ k_ws,
            const float* __restrict__ v_ws,
            const float* __restrict__ ln1_g, const float* __restrict__ ln1_b,
            const float* __restrict__ w1, const float* __restrict__ b1,
            const float* __restrict__ ln2_g, const float* __restrict__ ln2_b,
            const float* __restrict__ w2, const float* __restrict__ b2,
            const float* __restrict__ pe1, const float* __restrict__ pe2f,
            float* __restrict__ xout) {
  extern __shared__ char smem[];
  bf16_t* sW1  = (bf16_t*)smem;
  bf16_t* sW2  = sW1 + CD * CD;
  bf16_t* sPe2 = sW2 + CD * CD;
  bf16_t* sA   = sPe2 + CD * HIDD;
  float* sT1  = (float*)(sA + 16 * CD);
  float* sU0  = sT1 + 16 * CD;
  float* sU1  = sU0 + 16 * CD;
  float* sB1  = sU1 + 16 * CD;
  float* sB2  = sB1 + CD;
  float* sG1  = sB2 + CD;
  float* sE1  = sG1 + CD;
  float* sG2  = sE1 + CD;
  float* sE2  = sG2 + CD;
  float* sKi  = sE2 + CD;
  float* sRed = sKi + CD;
  float* sRed2= sRed + 128;
  float* sMean= sRed2 + 128;
  float* sInv = sMean + 16;
  float* sPi  = sInv + 16;

  int t = threadIdx.x;
  int n = blockIdx.x >> 7, i = blockIdx.x & 127;
  int row_i = n * PP + i;

  for (int idx = t; idx < CD * CD; idx += 128) {
    sW1[idx] = (bf16_t)w1[idx];
    sW2[idx] = (bf16_t)w2[idx];
  }
  for (int idx = t; idx < CD * HIDD; idx += 128) sPe2[idx] = (bf16_t)pe2f[idx];
  sB1[t] = b1[t]; sB2[t] = b2[t];
  sG1[t] = ln1_g[t]; sE1[t] = ln1_b[t];
  sG2[t] = ln2_g[t]; sE2[t] = ln2_b[t];
  sKi[t] = k_ws[row_i * CD + t];
  if (t < 4) sPi[t] = p[row_i * 4 + t];

  const size_t ubase = (size_t)row_i * PP * CD;
#if defined(HAVE_ASYNC_LDS)
  {  // kick off tile 0 DMA before anything else
    const char* src = (const char*)&U[ubase];
    char* dst = (char*)sU0;
#pragma unroll
    for (int qq = 0; qq < 4; ++qq) {
      int off = (qq * 128 + t) * 16;
      __builtin_amdgcn_global_load_async_to_lds_b128(
          (gp128_t)(src + off), (lp128_t)(dst + off), 0, 0);
    }
  }
#endif
  __syncthreads();

  const bool pad_i = (sPi[0] == 0.0f);
  float m_run = -3.0e38f, l_run = 0.f, acc_run = 0.f;
  float xvr[16];

  for (int jt = 0; jt < 8; ++jt) {
    int j0 = jt * 16;
#if defined(HAVE_ASYNC_LDS)
    if (jt < 7) {  // DMA next tile into the other buffer while we compute
      const char* src = (const char*)&U[ubase + (size_t)(j0 + 16) * CD];
      char* dst = (char*)((jt & 1) ? sU0 : sU1);
#pragma unroll
      for (int qq = 0; qq < 4; ++qq) {
        int off = (qq * 128 + t) * 16;
        __builtin_amdgcn_global_load_async_to_lds_b128(
            (gp128_t)(src + off), (lp128_t)(dst + off), 0, 0);
      }
    }
#endif
    // --- h1 = relu(ln(pij @ pe1.T)) : one thread per row (32-dim) ----------
    if (t < 16) {
      int rowj = n * PP + j0 + t;
      float pj0 = sPi[0] - p[rowj * 4 + 0];
      float pj1 = sPi[1] - p[rowj * 4 + 1];
      float pj2 = sPi[2] - p[rowj * 4 + 2];
      float pj3 = sPi[3] - p[rowj * 4 + 3];
      float h[HIDD]; float s = 0.f, s2 = 0.f;
#pragma unroll
      for (int o = 0; o < HIDD; ++o) {
        float a = pj0 * pe1[o * 4 + 0] + pj1 * pe1[o * 4 + 1] +
                  pj2 * pe1[o * 4 + 2] + pj3 * pe1[o * 4 + 3];
        h[o] = a; s += a; s2 += a * a;
      }
      float m = s * (1.f / 32.f);
      float var = s2 * (1.f / 32.f) - m * m;
      float iv = rsqrtf(var + 1e-5f);
#pragma unroll
      for (int o = 0; o < HIDD; ++o)
        sA[t * HIDD + o] = (bf16_t)fmaxf((h[o] - m) * iv, 0.f);
    }
    __syncthreads();
    // --- eqpe pre-LN = h1 @ pe2.T  (16x32x128 WMMA) ------------------------
    wmma_gemm<HIDD, CD>(sA, HIDD, sPe2, HIDD, sT1, CD);
    __syncthreads();
    ln_stats_16x128(sT1, sRed, sRed2, sMean, sInv);
    // --- combine with U, k_i, q_j, v_j (thread t owns channel c=t) ---------
#if defined(HAVE_ASYNC_LDS)
    if (jt < 7) { WAIT_ASYNC(4); } else { WAIT_ASYNC(0); }
    __syncthreads();  // make this tile's DMA visible to all waves
    const float* Ut = (jt & 1) ? sU1 : sU0;
#endif
#pragma unroll
    for (int r = 0; r < 16; ++r) {
      int idx = r * CD + t;
      float eq = fmaxf((sT1[idx] - sMean[r]) * sInv[r], 0.f);
      int rowj = n * PP + j0 + r;
#if defined(HAVE_ASYNC_LDS)
      float u = Ut[idx];
#else
      float u = U[ubase + (size_t)(j0 + r) * CD + t];
      if (jt < 7)
        __builtin_prefetch(&U[ubase + (size_t)(j0 + r + 16) * CD + t], 0, 1);
#endif
      float base = u + eq;
      xvr[r] = v_ws[rowj * CD + t] + base;
      sT1[idx] = sKi[t] - q_ws[rowj * CD + t] + base;
    }
    __syncthreads();
    // --- LN1(g,b)+relu -> bf16 A ------------------------------------------
    ln_stats_16x128(sT1, sRed, sRed2, sMean, sInv);
    for (int idx = t; idx < 16 * CD; idx += 128) {
      int r = idx >> 7, c = idx & 127;
      float y = (sT1[idx] - sMean[r]) * sInv[r] * sG1[c] + sE1[c];
      sA[idx] = (bf16_t)fmaxf(y, 0.f);
    }
    __syncthreads();
    wmma_gemm<CD, CD>(sA, CD, sW1, CD, sT1, CD);
    __syncthreads();
    for (int idx = t; idx < 16 * CD; idx += 128) sT1[idx] += sB1[idx & 127];
    __syncthreads();
    // --- LN2(g,b)+relu -> bf16 A ------------------------------------------
    ln_stats_16x128(sT1, sRed, sRed2, sMean, sInv);
    for (int idx = t; idx < 16 * CD; idx += 128) {
      int r = idx >> 7, c = idx & 127;
      float y = (sT1[idx] - sMean[r]) * sInv[r] * sG2[c] + sE2[c];
      sA[idx] = (bf16_t)fmaxf(y, 0.f);
    }
    __syncthreads();
    wmma_gemm<CD, CD>(sA, CD, sW2, CD, sT1, CD);
    __syncthreads();
    // --- online softmax over j (thread t owns channel c=t) ----------------
    float wv[16], tmax = -3.0e38f;
#pragma unroll
    for (int r = 0; r < 16; ++r) {
      float wvr = sT1[r * CD + t] + sB2[t];
      if (pad_i) wvr = NEGV;
      wv[r] = wvr;
      tmax = fmaxf(tmax, wvr);
    }
    float mnew = fmaxf(m_run, tmax);
    float sc = __expf(m_run - mnew);
    l_run *= sc; acc_run *= sc;
#pragma unroll
    for (int r = 0; r < 16; ++r) {
      float e = __expf(wv[r] - mnew);
      l_run += e;
      acc_run += e * xvr[r];
    }
    m_run = mnew;
    __syncthreads();
  }
  xout[row_i * CD + t] = acc_run / l_run;
}

// ---------------------------------------------------------------------------
// Kernel 3: a_i = x_out[i] @ e1[:, :128].T ; b_j = x_out[j] @ e1[:,128:256].T
// (decomposition of the K=257 pair GEMM: m_in@e1.T == a_i + b_j + e1[:,256]*norm)
// ---------------------------------------------------------------------------
__global__ void abproj_kernel(const float* __restrict__ xo,
                              const float* __restrict__ e1,
                              float* __restrict__ a_ws, float* __restrict__ b_ws) {
  __shared__ float sX[CD];
  int row = blockIdx.x, t = threadIdx.x;          // 64 threads
  sX[t] = xo[row * CD + t];
  sX[t + 64] = xo[row * CD + t + 64];
  __syncthreads();
  int o = t & 31;
  bool isB = t >= 32;
  const float* col = e1 + o * 257 + (isB ? 128 : 0);
  float s = 0.f;
#pragma unroll 8
  for (int c = 0; c < CD; ++c) s += sX[c] * col[c];
  if (isB) b_ws[row * 32 + o] = s;
  else     a_ws[row * 32 + o] = s;
}

// ---------------------------------------------------------------------------
// Kernel 4: per-(n,i) message stage -> s[j] -> softmax -> new_p.
// ---------------------------------------------------------------------------
__global__ void __launch_bounds__(128)
pair_kernel(const float* __restrict__ p, const float* __restrict__ a_ws,
            const float* __restrict__ b_ws, const float* __restrict__ e1,
            const float* __restrict__ e2f, const float* __restrict__ x1f,
            const float* __restrict__ x1b, const float* __restrict__ x2f,
            const float* __restrict__ x2b, float* __restrict__ newp) {
  extern __shared__ char smem[];
  bf16_t* sE2 = (bf16_t*)smem;                 // e2 (128,32)
  bf16_t* sX1 = sE2 + CD * HIDD;               // x1 (32,128)
  bf16_t* sA  = sX1 + HIDD * CD;               // 16x32
  bf16_t* sA2 = sA + 16 * HIDD;                // 16x128
  float* sT1  = (float*)(sA2 + 16 * CD);       // 16x128
  float* sO   = sT1 + 16 * CD;                 // 16x32
  float* sS   = sO + 16 * HIDD;                // 128
  float* sPij = sS + 128;                      // 128x4
  float* sAi  = sPij + 512;                    // 32
  float* sX2v = sAi + 32;                      // 32
  float* sRed = sX2v + 32;                     // 128
  float* sRed2= sRed + 128;                    // 128
  float* sMean= sRed2 + 128;                   // 16
  float* sInv = sMean + 16;                    // 16
  float* sPi  = sInv + 16;                     // 4
  float* sScal= sPi + 4;                       // 4

  int t = threadIdx.x;
  int n = blockIdx.x >> 7, i = blockIdx.x & 127;
  int row_i = n * PP + i;

  for (int idx = t; idx < CD * HIDD; idx += 128) {
    sE2[idx] = (bf16_t)e2f[idx];
    sX1[idx] = (bf16_t)x1f[idx];
  }
  if (t < 32) { sAi[t] = a_ws[row_i * 32 + t]; sX2v[t] = x2f[t]; }
  if (t < 4) sPi[t] = p[row_i * 4 + t];
  sRed[t] = (p[(n * PP + t) * 4] != 0.f) ? 1.f : 0.f;
  __syncthreads();
  if (t == 0) {
    float s = 0.f;
    for (int j = 0; j < PP; ++j) s += sRed[j];
    sScal[0] = s;
  }
  __syncthreads();
  const bool pad_i = (sPi[0] == 0.0f);
  const float reg = sScal[0];
  const float x2bb = x2b[0];

  for (int jt = 0; jt < 8; ++jt) {
    int j0 = jt * 16;
    if (t < 16) {
      int j = j0 + t, rowj = n * PP + j;
      float pj0 = sPi[0] - p[rowj * 4 + 0];
      float pj1 = sPi[1] - p[rowj * 4 + 1];
      float pj2 = sPi[2] - p[rowj * 4 + 2];
      float pj3 = sPi[3] - p[rowj * 4 + 3];
      sPij[j * 4 + 0] = pj0; sPij[j * 4 + 1] = pj1;
      sPij[j * 4 + 2] = pj2; sPij[j * 4 + 3] = pj3;
      float nrm = pj3 * pj3 - (pj0 * pj0 + pj1 * pj1 + pj2 * pj2);
      if (pad_i) nrm = 0.f;
#pragma unroll
      for (int o = 0; o < HIDD; ++o) {
        float h = sAi[o] + b_ws[rowj * 32 + o] + e1[o * 257 + 256] * nrm;
        sA[t * HIDD + o] = (bf16_t)fmaxf(h, 0.f);
      }
    }
    __syncthreads();
    wmma_gemm<HIDD, CD>(sA, HIDD, sE2, HIDD, sT1, CD);    // mij pre-relu
    __syncthreads();
    for (int idx = t; idx < 16 * CD; idx += 128) {
      float v = fmaxf(sT1[idx], 0.f);
      if (pad_i) v = NEGV;
      sT1[idx] = v;
    }
    __syncthreads();
    ln_stats_16x128(sT1, sRed, sRed2, sMean, sInv);
    for (int idx = t; idx < 16 * CD; idx += 128) {
      int r = idx >> 7;
      sA2[idx] = (bf16_t)((sT1[idx] - sMean[r]) * sInv[r]);
    }
    __syncthreads();
    wmma_gemm<CD, HIDD>(sA2, CD, sX1, CD, sO, HIDD);       // @ x1.T
    __syncthreads();
    if (t < 16) {
      float s = x2bb;
#pragma unroll
      for (int o = 0; o < HIDD; ++o) {
        float hh = fmaxf(sO[t * HIDD + o] + x1b[o], 0.f);
        s += hh * sX2v[o];
      }
      sS[j0 + t] = fmaxf(s, 0.f);
    }
    __syncthreads();
  }
  // softmax over j, then weighted pij sum
  sRed[t] = sS[t]; __syncthreads();
  for (int off = 64; off; off >>= 1) {
    if (t < off) sRed[t] = fmaxf(sRed[t], sRed[t + off]);
    __syncthreads();
  }
  float smax = sRed[0]; __syncthreads();
  float e = __expf(sS[t] - smax);
  sRed[t] = e; __syncthreads();
  for (int off = 64; off; off >>= 1) {
    if (t < off) sRed[t] += sRed[t + off];
    __syncthreads();
  }
  float tot = sRed[0]; __syncthreads();
  float pw = e / tot;
  float npv[4];
#pragma unroll
  for (int d = 0; d < 4; ++d) {
    sRed[t] = pw * sPij[t * 4 + d]; __syncthreads();
    for (int off = 64; off; off >>= 1) {
      if (t < off) sRed[t] += sRed[t + off];
      __syncthreads();
    }
    npv[d] = sRed[0]; __syncthreads();
  }
  if (t < 4) {
    float pv = sPi[t];
    newp[row_i * 4 + t] = (pv == 0.f) ? 0.f : pv + npv[t] / reg;
  }
}

// ---------------------------------------------------------------------------
extern "C" void kernel_launch(void* const* d_in, const int* in_sizes, int n_in,
                              void* d_out, int out_size, void* d_ws, size_t ws_size,
                              hipStream_t stream) {
  const float* x     = (const float*)d_in[0];
  const float* p     = (const float*)d_in[1];
  const float* U     = (const float*)d_in[2];
  const float* wq    = (const float*)d_in[3];
  const float* bq    = (const float*)d_in[4];
  const float* wk    = (const float*)d_in[5];
  const float* bk    = (const float*)d_in[6];
  const float* wv    = (const float*)d_in[7];
  const float* bv    = (const float*)d_in[8];
  const float* ln1_g = (const float*)d_in[9];
  const float* ln1_b = (const float*)d_in[10];
  const float* w1    = (const float*)d_in[11];
  const float* b1    = (const float*)d_in[12];
  const float* ln2_g = (const float*)d_in[13];
  const float* ln2_b = (const float*)d_in[14];
  const float* w2    = (const float*)d_in[15];
  const float* b2    = (const float*)d_in[16];
  const float* pe1   = (const float*)d_in[17];
  const float* pe2   = (const float*)d_in[18];
  const float* e1    = (const float*)d_in[19];
  const float* e2    = (const float*)d_in[20];
  const float* x1    = (const float*)d_in[21];
  const float* x1b   = (const float*)d_in[22];
  const float* x2    = (const float*)d_in[23];
  const float* x2b   = (const float*)d_in[24];

  float* ws   = (float*)d_ws;               // needs 917504 floats (~3.5 MB)
  float* q_ws = ws;
  float* k_ws = ws + (size_t)NB * PP * CD;
  float* v_ws = k_ws + (size_t)NB * PP * CD;
  float* a_ws = v_ws + (size_t)NB * PP * CD;
  float* b_ws = a_ws + (size_t)NB * PP * HIDD;

  float* xout = (float*)d_out;
  float* newp = xout + (size_t)NB * PP * CD;

  const int nRows = NB * PP;                // 2048
  const size_t smemA = 110592;              // ~107 KB used
  const size_t smemB = 40960;               // ~36 KB used
  (void)hipFuncSetAttribute((const void*)main_kernel,
      hipFuncAttributeMaxDynamicSharedMemorySize, (int)smemA);
  (void)hipFuncSetAttribute((const void*)pair_kernel,
      hipFuncAttributeMaxDynamicSharedMemorySize, (int)smemB);

  qkv_kernel<<<nRows, 128, 0, stream>>>(x, wq, bq, wk, bk, wv, bv,
                                        q_ws, k_ws, v_ws);
  main_kernel<<<nRows, 128, smemA, stream>>>(p, U, q_ws, k_ws, v_ws,
      ln1_g, ln1_b, w1, b1, ln2_g, ln2_b, w2, b2, pe1, pe2, xout);
  abproj_kernel<<<nRows, 64, 0, stream>>>(xout, e1, a_ws, b_ws);
  pair_kernel<<<nRows, 128, smemB, stream>>>(p, a_ws, b_ws, e1, e2,
      x1, x1b, x2, x2b, newp);
}
